// LinearAttention_80874234184125
// MI455X (gfx1250) — matrix operand order repaired and checked by reading, not compiled, not run on hardware
//
#include <hip/hip_runtime.h>
#include <hip/hip_bf16.h>
#include <math.h>

// ---------------------------------------------------------------------------
// Gated linear attention, chunked-parallel form, f16 WMMA (fp32 accumulate).
//   T=2048, D=1024, H=16 heads, dh=64, chunk C=64.
//   GEMMs use the Tensor Data Mover (tensor_load_to_lds + s_wait_tensorcnt)
//   to double-buffer 64x64 f16 tiles in LDS; 8 waves/WG, 4x A/B reuse.
// ---------------------------------------------------------------------------

typedef __attribute__((ext_vector_type(16))) _Float16 v16h;
typedef __attribute__((ext_vector_type(8)))  _Float16 v8h;
typedef __attribute__((ext_vector_type(8)))  float    v8f;
typedef unsigned int u32x4 __attribute__((ext_vector_type(4)));
typedef int          i32x8 __attribute__((ext_vector_type(8)));
typedef int          i32x4 __attribute__((ext_vector_type(4)));

#define T_LEN 2048
#define D_DIM 1024
#define N_HEADS 16
#define DH 64
#define CHUNK 64

#define MB 64
#define NB 64
#define KB 64

#if __has_builtin(__builtin_amdgcn_tensor_load_to_lds) && __has_builtin(__builtin_amdgcn_s_wait_tensorcnt)
#define HAVE_TDM 1
#else
#define HAVE_TDM 0
#endif

// ---- WMMA fragment helpers (wave32, layouts per CDNA5 ISA 7.12.2) ----------

// A matrix 16x32 f16: lane (h,r): elems 0..7 = K(h*8..), elems 8..15 = K(16+h*8..)
__device__ __forceinline__ v16h load_a_frag(const _Float16* M, int ld, int m0, int k0, int lane) {
    const int r = lane & 15, hh = lane >> 4;
    const _Float16* p = M + (size_t)(m0 + r) * ld + k0 + hh * 8;
    v8h lo = *(const v8h*)(p);
    v8h hi = *(const v8h*)(p + 16);
    v16h out;
    #pragma unroll
    for (int i = 0; i < 8; ++i) { out[i] = lo[i]; out[i + 8] = hi[i]; }
    return out;
}

// B matrix 32x16 f16 (from row-major Mat[n][k]): lane (h,r): elems 0..7 = K(h*8..), elems 8..15 = K(16+h*8..) of col n0+r
__device__ __forceinline__ v16h load_b_frag(const _Float16* M, int ld, int n0, int k0, int lane) {
    const int r = lane & 15, hh = lane >> 4;
    const _Float16* p = M + (size_t)(n0 + r) * ld + k0 + hh * 8;
    v8h lo = *(const v8h*)(p);
    v8h hi = *(const v8h*)(p + 16);
    v16h out;
    #pragma unroll
    for (int i = 0; i < 8; ++i) { out[i] = lo[i]; out[i + 8] = hi[i]; }
    return out;
}

__device__ __forceinline__ v8f wmma_f16(v16h a, v16h b, v8f c) {
    return __builtin_amdgcn_wmma_f32_16x16x32_f16(false, a, false, b, (short)0, c, false, false);
}

// ---- Tensor Data Mover helpers --------------------------------------------

__device__ __forceinline__ unsigned lds_offset_of(const void* p) {
    // flat LDS aperture: low 32 bits of the generic address are the LDS byte offset
    return (unsigned)(unsigned long long)(uintptr_t)p;
}

#if HAVE_TDM
// Issue a TDM load of a 2D tile: KB elements (dim0) x 64 rows (dim1), f16,
// from row-major matrix gbase[ld] starting at (row0, k0), into LDS at lds_off.
__device__ __forceinline__ void tdm_load_tile(const _Float16* gbase, int ld, int nrows,
                                              int row0, int k0, unsigned lds_off) {
    unsigned long long ga = (unsigned long long)(uintptr_t)(gbase + (size_t)row0 * ld + k0);
    u32x4 g0;
    g0[0] = 1u;                                   // count=1 (valid), user-mode
    g0[1] = lds_off;                              // lds_addr
    g0[2] = (unsigned)(ga & 0xFFFFFFFFull);       // global_addr[31:0]
    g0[3] = (unsigned)(ga >> 32) | (2u << 30);    // global_addr[56:32] | type=2 (image)
    i32x8 g1;
    g1[0] = (1 << 16);                            // wg_mask=0, data_size=1 -> 2 bytes
    g1[1] = (int)(((unsigned)ld & 0xFFFFu) << 16);            // tensor_dim0[15:0] @ bits 63:48
    g1[2] = (int)((((unsigned)ld >> 16) & 0xFFFFu)            // tensor_dim0[31:16]
                | (((unsigned)nrows & 0xFFFFu) << 16));       // tensor_dim1[15:0]
    g1[3] = (int)((((unsigned)nrows >> 16) & 0xFFFFu)         // tensor_dim1[31:16]
                | ((unsigned)KB << 16));                      // tile_dim0 = KB
    g1[4] = 64;                                   // tile_dim1 = 64 rows, tile_dim2 = 0
    g1[5] = ld;                                   // tensor_dim0_stride[31:0]
    g1[6] = 0;                                    // stride hi / tensor_dim1_stride lo
    g1[7] = 0;
    // 6-arg flavor: groups 2/3 (unused for 2D) + extra group slot + cache policy
    __builtin_amdgcn_tensor_load_to_lds(g0, g1, (i32x4)0, (i32x4)0, (i32x8)0, 0);
}
#endif

// ---- 1) fp32 -> f16 conversion --------------------------------------------

__global__ void cvt_f32_to_f16_kernel(const float* __restrict__ in,
                                      _Float16* __restrict__ out, int n) {
    int i = blockIdx.x * 256 + threadIdx.x;
    if (i < n) out[i] = (_Float16)in[i];
}

// ---- 2/5) NT GEMM: C[M][N] = A[M][K] * B[N][K]^T --------------------------
// 64x64 block tile, 8 waves, TDM double-buffered LDS staging.

__global__ __launch_bounds__(256)
void gemm_nt_wmma_kernel(const _Float16* __restrict__ A,
                         const _Float16* __restrict__ B,
                         _Float16* __restrict__ C16,
                         float* __restrict__ C32,
                         int M, int N, int K) {
    __shared__ __attribute__((aligned(128))) _Float16 Ash[2][MB * KB];
    __shared__ __attribute__((aligned(128))) _Float16 Bsh[2][NB * KB];

    const int tid = threadIdx.x;
    const int lane = tid & 31;
    const int wave = tid >> 5;
    const int m0 = blockIdx.y * MB;
    const int n0 = blockIdx.x * NB;

    v8f acc[2] = {{}, {}};

#if HAVE_TDM
    // prologue: stage 0
    if (wave == 0) {
        tdm_load_tile(A, K, M, m0, 0, lds_offset_of(&Ash[0][0]));
        tdm_load_tile(B, K, N, n0, 0, lds_offset_of(&Bsh[0][0]));
    }
    for (int kk = 0; kk < K; kk += KB) {
        const int s = (kk / KB) & 1;
        const bool has_next = (kk + KB) < K;
        if (wave == 0) {
            if (has_next) {
                tdm_load_tile(A, K, M, m0, kk + KB, lds_offset_of(&Ash[s ^ 1][0]));
                tdm_load_tile(B, K, N, n0, kk + KB, lds_offset_of(&Bsh[s ^ 1][0]));
                __builtin_amdgcn_s_wait_tensorcnt((short)2);   // current stage complete
            } else {
                __builtin_amdgcn_s_wait_tensorcnt((short)0);
            }
        }
        __syncthreads();   // stage s visible to all waves
        #pragma unroll
        for (int t = 0; t < 2; ++t) {
            const int tile = wave + 8 * t;
            const int ti = tile >> 2, tj = tile & 3;
            #pragma unroll
            for (int k0 = 0; k0 < KB; k0 += 32) {
                v16h a = load_a_frag(&Ash[s][0], KB, ti * 16, k0, lane);
                v16h b = load_b_frag(&Bsh[s][0], KB, tj * 16, k0, lane);
                acc[t] = wmma_f16(a, b, acc[t]);
            }
        }
        __syncthreads();   // all reads of stage s done before it is overwritten
    }
#else
    // fallback: cooperative copy, single buffer
    for (int kk = 0; kk < K; kk += KB) {
        for (int i = tid; i < MB * KB; i += 256) {
            int rr = i / KB, cc = i % KB;
            Ash[0][i] = A[(size_t)(m0 + rr) * K + kk + cc];
            Bsh[0][i] = B[(size_t)(n0 + rr) * K + kk + cc];
        }
        __syncthreads();
        #pragma unroll
        for (int t = 0; t < 2; ++t) {
            const int tile = wave + 8 * t;
            const int ti = tile >> 2, tj = tile & 3;
            #pragma unroll
            for (int k0 = 0; k0 < KB; k0 += 32) {
                v16h a = load_a_frag(&Ash[0][0], KB, ti * 16, k0, lane);
                v16h b = load_b_frag(&Bsh[0][0], KB, tj * 16, k0, lane);
                acc[t] = wmma_f16(a, b, acc[t]);
            }
        }
        __syncthreads();
    }
#endif

    const int r = lane & 15, hh = lane >> 4;
    #pragma unroll
    for (int t = 0; t < 2; ++t) {
        const int tile = wave + 8 * t;
        const int ti = tile >> 2, tj = tile & 3;
        if (C32) {
            #pragma unroll
            for (int i = 0; i < 8; ++i)
                C32[(size_t)(m0 + ti * 16 + hh * 8 + i) * N + n0 + tj * 16 + r] = acc[t][i];
        } else {
            #pragma unroll
            for (int i = 0; i < 8; ++i)
                C16[(size_t)(m0 + ti * 16 + hh * 8 + i) * N + n0 + tj * 16 + r] = (_Float16)acc[t][i];
        }
    }
}

// ---- 3) activations --------------------------------------------------------

__device__ __forceinline__ float softplus_f(float x) {
    return (x > 20.f) ? x : logf(1.f + expf(x));
}

__global__ void act_kernel(_Float16* __restrict__ q, _Float16* __restrict__ k,
                           const _Float16* __restrict__ g,
                           const float* __restrict__ bg, int n) {
    int i = blockIdx.x * 256 + threadIdx.x;
    if (i >= n) return;
    int col = i & (D_DIM - 1);
    float qv = (float)q[i];
    float kv = (float)k[i];
    float gv = (float)g[i] + bg[col];
    float sg = 1.f / (1.f + expf(-gv));
    q[i] = (_Float16)(softplus_f(qv) * (sg + 1e-6f));
    k[i] = (_Float16)softplus_f(kv);
}

// ---- 4) chunked linear-attention recurrence, one block per head ------------

__global__ __launch_bounds__(256)
void chunk_attn_kernel(const _Float16* __restrict__ qf,
                       const _Float16* __restrict__ kf,
                       const _Float16* __restrict__ v,
                       _Float16* __restrict__ outpre,
                       float* __restrict__ state_out,  // [H][DH][DH]
                       float* __restrict__ z_out) {    // [H][DH]
    __shared__ __attribute__((aligned(32))) float    ST[DH * DH];     // S^T: [e][d]
    __shared__ __attribute__((aligned(32))) _Float16 ST16[DH * DH];   // f16 snapshot
    __shared__ __attribute__((aligned(32))) _Float16 A16[CHUNK * CHUNK];
    __shared__ __attribute__((aligned(32))) _Float16 vT[DH * CHUNK];  // [e][t]
    __shared__ __attribute__((aligned(32))) _Float16 kT[DH * CHUNK];  // [d][t]
    __shared__ float den[CHUNK];
    __shared__ float zsh[DH];

    const int h = blockIdx.x;
    const int col0 = h * DH;
    const int tid = threadIdx.x;
    const int lane = tid & 31;
    const int wave = tid >> 5;

    for (int i = tid; i < DH * DH; i += 256) ST[i] = 0.f;
    if (tid < DH) zsh[tid] = 0.f;
    __syncthreads();

    for (int c0 = 0; c0 < T_LEN; c0 += CHUNK) {
        // (a) transposed chunk copies + f16 state snapshot
        for (int i = tid; i < DH * CHUNK; i += 256) {
            int e = i >> 6, t = i & 63;
            vT[e * CHUNK + t] = v[(size_t)(c0 + t) * D_DIM + col0 + e];
            kT[e * CHUNK + t] = kf[(size_t)(c0 + t) * D_DIM + col0 + e];
            ST16[i] = (_Float16)ST[i];
        }
        __syncthreads();

        // (b) A_intra = tril(qf_c @ kf_c^T)
        for (int tile = wave; tile < 16; tile += 8) {
            int ti = tile >> 2, tj = tile & 3;
            v8f c = {};
            for (int k0 = 0; k0 < DH; k0 += 32) {
                v16h a = load_a_frag(qf, D_DIM, c0 + ti * 16, col0 + k0, lane);
                v16h b = load_b_frag(kf, D_DIM, c0 + tj * 16, col0 + k0, lane);
                c = wmma_f16(a, b, c);
            }
            const int r = lane & 15, hh = lane >> 4;
            #pragma unroll
            for (int i = 0; i < 8; ++i) {
                int tr = ti * 16 + hh * 8 + i, tc = tj * 16 + r;
                A16[tr * CHUNK + tc] = (tr >= tc) ? (_Float16)c[i] : (_Float16)0.f;
            }
        }
        __syncthreads();

        // (c) den[t] = rowsum(A_intra) + qf_t . z_prev + 1e-6
        if (tid < CHUNK) {
            int t = tid;
            float s = 0.f;
            for (int j = 0; j < CHUNK; ++j) s += (float)A16[t * CHUNK + j];
            const _Float16* qrow = qf + (size_t)(c0 + t) * D_DIM + col0;
            for (int d = 0; d < DH; ++d) s += (float)qrow[d] * zsh[d];
            den[t] = s + 1e-6f;
        }
        __syncthreads();

        // (d) num = qf_c @ S + A_intra @ v_c ; out = num/den
        for (int tile = wave; tile < 16; tile += 8) {
            int ti = tile >> 2, ej = tile & 3;
            v8f c = {};
            for (int k0 = 0; k0 < DH; k0 += 32) {
                v16h a = load_a_frag(qf, D_DIM, c0 + ti * 16, col0 + k0, lane);
                v16h b = load_b_frag(ST16, DH, ej * 16, k0, lane);
                c = wmma_f16(a, b, c);
            }
            for (int k0 = 0; k0 < CHUNK; k0 += 32) {
                v16h a = load_a_frag(A16, CHUNK, ti * 16, k0, lane);
                v16h b = load_b_frag(vT, CHUNK, ej * 16, k0, lane);
                c = wmma_f16(a, b, c);
            }
            const int r = lane & 15, hh = lane >> 4;
            #pragma unroll
            for (int i = 0; i < 8; ++i) {
                int tr = ti * 16 + hh * 8 + i, ec = ej * 16 + r;
                outpre[(size_t)(c0 + tr) * D_DIM + col0 + ec] = (_Float16)(c[i] / den[tr]);
            }
        }

        // (e) state update: ST[e][d] += sum_t v[t][e]*kf[t][d]
        for (int tile = wave; tile < 16; tile += 8) {
            int ei = tile >> 2, dj = tile & 3;
            const int r = lane & 15, hh = lane >> 4;
            v8f c;
            #pragma unroll
            for (int i = 0; i < 8; ++i)
                c[i] = ST[(ei * 16 + hh * 8 + i) * DH + dj * 16 + r];
            for (int k0 = 0; k0 < CHUNK; k0 += 32) {
                v16h a = load_a_frag(vT, CHUNK, ei * 16, k0, lane);
                v16h b = load_b_frag(kT, CHUNK, dj * 16, k0, lane);
                c = wmma_f16(a, b, c);
            }
            #pragma unroll
            for (int i = 0; i < 8; ++i)
                ST[(ei * 16 + hh * 8 + i) * DH + dj * 16 + r] = c[i];
        }

        // (f) z update
        if (tid < DH) {
            int d = tid;
            float s = zsh[d];
            for (int t = 0; t < CHUNK; ++t) s += (float)kT[d * CHUNK + t];
            zsh[d] = s;
        }
        __syncthreads();
    }

    // tail outputs: kv_cum[-1][d][e] = ST[e][d] ; z_cum[-1]
    for (int i = tid; i < DH * DH; i += 256) {
        int d = i >> 6, e = i & 63;
        state_out[(size_t)h * DH * DH + i] = ST[e * DH + d];
    }
    if (tid < DH) z_out[h * DH + tid] = zsh[tid];
}

// ---------------------------------------------------------------------------

extern "C" void kernel_launch(void* const* d_in, const int* in_sizes, int n_in,
                              void* d_out, int out_size, void* d_ws, size_t ws_size,
                              hipStream_t stream) {
    const float* x  = (const float*)d_in[0];
    const float* Wq = (const float*)d_in[1];
    const float* Wk = (const float*)d_in[2];
    const float* Wv = (const float*)d_in[3];
    const float* Wo = (const float*)d_in[4];
    const float* Wg = (const float*)d_in[5];
    const float* bg = (const float*)d_in[6];

    const size_t TD = (size_t)T_LEN * D_DIM;   // 2,097,152
    const size_t DD = (size_t)D_DIM * D_DIM;   // 1,048,576

    _Float16* ws = (_Float16*)d_ws;
    _Float16* x16  = ws;            // TD
    _Float16* Wq16 = x16  + TD;     // DD
    _Float16* Wk16 = Wq16 + DD;
    _Float16* Wv16 = Wk16 + DD;
    _Float16* Wg16 = Wv16 + DD;
    _Float16* Wo16 = Wg16 + DD;
    _Float16* q16  = Wo16 + DD;     // TD  (becomes qf in place)
    _Float16* k16  = q16  + TD;     // TD  (becomes kf in place)
    _Float16* v16  = k16  + TD;     // TD
    _Float16* g16  = v16  + TD;     // TD
    _Float16* op16 = g16  + TD;     // TD  (pre-Wo output, f16)

    float* out32     = (float*)d_out;                  // [T][D]
    float* state_out = out32 + TD;                     // [H][DH][DH]
    float* z_out     = state_out + (size_t)N_HEADS * DH * DH;  // [H][DH]

    // 1) convert inputs to f16
    cvt_f32_to_f16_kernel<<<(TD + 255) / 256, 256, 0, stream>>>(x,  x16,  (int)TD);
    cvt_f32_to_f16_kernel<<<(DD + 255) / 256, 256, 0, stream>>>(Wq, Wq16, (int)DD);
    cvt_f32_to_f16_kernel<<<(DD + 255) / 256, 256, 0, stream>>>(Wk, Wk16, (int)DD);
    cvt_f32_to_f16_kernel<<<(DD + 255) / 256, 256, 0, stream>>>(Wv, Wv16, (int)DD);
    cvt_f32_to_f16_kernel<<<(DD + 255) / 256, 256, 0, stream>>>(Wg, Wg16, (int)DD);
    cvt_f32_to_f16_kernel<<<(DD + 255) / 256, 256, 0, stream>>>(Wo, Wo16, (int)DD);

    // 2) projections: [2048x1024] = x16 @ W^T  (TDM-staged WMMA GEMM)
    dim3 ggrid(D_DIM / NB, T_LEN / MB);   // (16, 32)
    gemm_nt_wmma_kernel<<<ggrid, 256, 0, stream>>>(x16, Wq16, q16, nullptr, T_LEN, D_DIM, D_DIM);
    gemm_nt_wmma_kernel<<<ggrid, 256, 0, stream>>>(x16, Wk16, k16, nullptr, T_LEN, D_DIM, D_DIM);
    gemm_nt_wmma_kernel<<<ggrid, 256, 0, stream>>>(x16, Wv16, v16, nullptr, T_LEN, D_DIM, D_DIM);
    gemm_nt_wmma_kernel<<<ggrid, 256, 0, stream>>>(x16, Wg16, g16, nullptr, T_LEN, D_DIM, D_DIM);

    // 3) activations in place (q16 -> qf, k16 -> kf)
    act_kernel<<<(TD + 255) / 256, 256, 0, stream>>>(q16, k16, g16, bg, (int)TD);

    // 4) chunked recurrence per head; writes op16 + final state/z
    chunk_attn_kernel<<<N_HEADS, 256, 0, stream>>>(q16, k16, v16, op16, state_out, z_out);

    // 5) out = op16 @ Wo^T  (fp32 store straight to d_out)
    gemm_nt_wmma_kernel<<<ggrid, 256, 0, stream>>>(op16, Wo16, nullptr, out32, T_LEN, D_DIM, D_DIM);
}
